// GlobalCoC_1434519077366
// MI455X (gfx1250) — matrix operand order repaired and checked by reading, not compiled, not run on hardware
//
#include <hip/hip_runtime.h>
#include <cmath>

typedef __bf16 bhalf;
typedef __attribute__((ext_vector_type(16))) __bf16 v16bh;
typedef __attribute__((ext_vector_type(8)))  float  v8f;
typedef int v4i __attribute__((vector_size(16)));   // matches builtin param type

#define NBATCH 8
#define HDIM   52
#define LTOK   2704      // 52*52
#define STOK   676       // 26*26
#define CCH    256
#define NL     (NBATCH*LTOK)   // 21632
#define NS     (NBATCH*STOK)   // 5408

__device__ __forceinline__ float gelu_f(float x) {
  return 0.5f * x * (1.0f + erff(x * 0.70710678118654752f));
}

// ---------------- async global->LDS staging (CDNA5 path), guarded ----------------
#if __has_builtin(__builtin_amdgcn_global_load_async_to_lds_b128)
#define HAVE_ASYNC_LDS 1
__device__ __forceinline__ void async_ld16(const void* g, unsigned lds_off) {
  __builtin_amdgcn_global_load_async_to_lds_b128(
      (__attribute__((address_space(1))) v4i*)(unsigned long long)(uintptr_t)g,
      (__attribute__((address_space(3))) v4i*)(unsigned long long)lds_off, 0, 0);
}
__device__ __forceinline__ void async_wait0() {
#if __has_builtin(__builtin_amdgcn_s_wait_asynccnt)
  __builtin_amdgcn_s_wait_asynccnt(0);
#else
  asm volatile("s_wait_asynccnt 0x0" ::: "memory");
#endif
}
__device__ __forceinline__ void async_wait3() {   // keep newest 3 (next tile) in flight
#if __has_builtin(__builtin_amdgcn_s_wait_asynccnt)
  __builtin_amdgcn_s_wait_asynccnt(3);
#else
  asm volatile("s_wait_asynccnt 0x3" ::: "memory");
#endif
}
__device__ __forceinline__ unsigned lds_off_of(const void* p) {
  return (unsigned)(uintptr_t)p;
}
#endif

// ---------------------------------------------------------------- utilities
__global__ void k_zero(float* p, int n) {
  int i = blockIdx.x * blockDim.x + threadIdx.x;
  if (i < n) p[i] = 0.0f;
}

__global__ void k_f32_to_bf16(const float* __restrict__ src, bhalf* __restrict__ dst, int n) {
  int i = blockIdx.x * blockDim.x + threadIdx.x;
  if (i < n) dst[i] = (bhalf)src[i];
}

// gb_conv_w: OIHW (256,512,3,3) -> Wp[tap][cin(512)][cout(256)] bf16
__global__ void k_pack_conv_w(const float* __restrict__ w, bhalf* __restrict__ wp) {
  int i = blockIdx.x * blockDim.x + threadIdx.x;
  const int total = 9 * 512 * 256;
  if (i >= total) return;
  int o   = i & 255;
  int t   = i >> 8;
  int cin = t & 511;
  int tap = t >> 9;
  wp[i] = (bhalf)w[((size_t)o * 512 + cin) * 9 + tap];
}

// ------------------------------------------------------- merge: concat + upsample
__global__ __launch_bounds__(256) void k_merge_cat(const float* __restrict__ x,
                                                   const float* __restrict__ cen,
                                                   bhalf* __restrict__ cat) {
  const int n = blockIdx.y, l = blockIdx.x, c = threadIdx.x;
  const size_t ro = ((size_t)n * LTOK + l) * 512;
  cat[ro + c] = (bhalf)x[((size_t)n * LTOK + l) * CCH + c];
  const int oy = l / HDIM, ox = l % HDIM;
  const float fy = oy * (25.0f / 51.0f);
  const float fx = ox * (25.0f / 51.0f);
  int y0 = (int)fy, x0 = (int)fx;
  int y1 = min(y0 + 1, 25), x1 = min(x0 + 1, 25);
  float dy = fy - (float)y0, dx = fx - (float)x0;
  const float* cb = cen + (size_t)n * STOK * CCH + c;
  float v00 = cb[(size_t)(y0 * 26 + x0) * CCH];
  float v01 = cb[(size_t)(y0 * 26 + x1) * CCH];
  float v10 = cb[(size_t)(y1 * 26 + x0) * CCH];
  float v11 = cb[(size_t)(y1 * 26 + x1) * CCH];
  float v = (v00 * (1.f - dx) + v01 * dx) * (1.f - dy) + (v10 * (1.f - dx) + v11 * dx) * dy;
  cat[ro + 256 + c] = (bhalf)v;
}

// ------------------------------------------------------------------- WMMA GEMM
// C[M,N] = A[M,K] @ B[K,N] (+bias) (optional exact GELU); A,B bf16 row-major.
// 64x128 tile, 8 waves (4M x 2N), wave = 16x64 = 4 WMMA; double-buffered async LDS.
__global__ __launch_bounds__(256) void k_gemm(const bhalf* __restrict__ A,
                                              const bhalf* __restrict__ B,
                                              const float* __restrict__ bias,
                                              float* __restrict__ outF,
                                              bhalf* __restrict__ outB,
                                              int M, int N, int K, int doGelu) {
  __shared__ bhalf sA[2][64][40];
  __shared__ bhalf sB[2][32][136];
  const int tid = threadIdx.x;
  const int wave = tid >> 5, lane = tid & 31;
  const int m0 = blockIdx.x * 64, n0 = blockIdx.y * 128;
  const int tm = (wave & 3) * 16, tn = (wave >> 2) * 64;
  v8f acc[4];
#pragma unroll
  for (int j = 0; j < 4; ++j) acc[j] = (v8f){0.f, 0.f, 0.f, 0.f, 0.f, 0.f, 0.f, 0.f};
  const int aRow = tm + (lane & 15);
  const int aK0  = (lane < 16) ? 0 : 8;               // ISA 16-bit A lane layout
  const int lar = tid >> 2, lac = (tid & 3) * 8;      // A staging: 64x32, 8 bf16/thread
  const int lbr0 = tid >> 4, lbc0 = (tid & 15) * 8;   // B staging rows 0..15
  const int lbr1 = lbr0 + 16;                         // B staging rows 16..31
  // signed row index: ragged-M tiles read a few KB past A (ws-interior), masked at store
  const bhalf* aPtr  = A + (long)(m0 + lar) * K + lac;
  const bhalf* bPtr0 = B + (size_t)lbr0 * N + n0 + lbc0;
  const bhalf* bPtr1 = B + (size_t)lbr1 * N + n0 + lbc0;
#ifdef HAVE_ASYNC_LDS
  const unsigned loA[2]  = {lds_off_of(&sA[0][lar][lac]),  lds_off_of(&sA[1][lar][lac])};
  const unsigned loB0[2] = {lds_off_of(&sB[0][lbr0][lbc0]), lds_off_of(&sB[1][lbr0][lbc0])};
  const unsigned loB1[2] = {lds_off_of(&sB[0][lbr1][lbc0]), lds_off_of(&sB[1][lbr1][lbc0])};
  // prologue: stage k-tile 0 into buffer 0
  async_ld16(aPtr, loA[0]);
  async_ld16(bPtr0, loB0[0]);
  async_ld16(bPtr1, loB1[0]);
  int cur = 0;
  for (int kb = 0; kb < K; kb += 32) {
    const int nxt = cur ^ 1;
    const bool more = (kb + 32) < K;
    if (more) {  // overlap: issue next tile before waiting on current
      async_ld16(aPtr + kb + 32, loA[nxt]);
      async_ld16(bPtr0 + (size_t)(kb + 32) * N, loB0[nxt]);
      async_ld16(bPtr1 + (size_t)(kb + 32) * N, loB1[nxt]);
    }
    if (more) async_wait3(); else async_wait0();  // in-order completion => current landed
    __syncthreads();
    const bhalf (*pA)[40]  = sA[cur];
    const bhalf (*pB)[136] = sB[cur];
    v16bh af;
#pragma unroll
    for (int i = 0; i < 8; ++i) {
      af[i]     = pA[aRow][aK0 + i];
      af[8 + i] = pA[aRow][aK0 + 16 + i];
    }
#pragma unroll
    for (int j = 0; j < 4; ++j) {
      v16bh bf;
#pragma unroll
      for (int i = 0; i < 8; ++i) {
        bf[i]     = pB[aK0 + i][tn + j * 16 + (lane & 15)];
        bf[8 + i] = pB[aK0 + 16 + i][tn + j * 16 + (lane & 15)];
      }
      acc[j] = __builtin_amdgcn_wmma_f32_16x16x32_bf16(false, af, false, bf, (short)0,
                                                       acc[j], false, false);
    }
    __syncthreads();  // all reads done before buffer becomes an async destination
    cur = nxt;
  }
#else
  const int garOk = (m0 + lar) < M;
  for (int kb = 0; kb < K; kb += 32) {
    __syncthreads();
    {
      uint4 va = make_uint4(0, 0, 0, 0);
      if (garOk) va = *(const uint4*)(aPtr + kb);
      *(uint4*)&sA[0][lar][lac] = va;
      *(uint4*)&sB[0][lbr0][lbc0] = *(const uint4*)(bPtr0 + (size_t)kb * N);
      *(uint4*)&sB[0][lbr1][lbc0] = *(const uint4*)(bPtr1 + (size_t)kb * N);
    }
    if (kb + 32 < K) {
      __builtin_prefetch(aPtr + kb + 32, 0, 1);
      __builtin_prefetch(bPtr0 + (size_t)(kb + 32) * N, 0, 1);
    }
    __syncthreads();
    v16bh af;
#pragma unroll
    for (int i = 0; i < 8; ++i) {
      af[i]     = sA[0][aRow][aK0 + i];
      af[8 + i] = sA[0][aRow][aK0 + 16 + i];
    }
#pragma unroll
    for (int j = 0; j < 4; ++j) {
      v16bh bf;
#pragma unroll
      for (int i = 0; i < 8; ++i) {
        bf[i]     = sB[0][aK0 + i][tn + j * 16 + (lane & 15)];
        bf[8 + i] = sB[0][aK0 + 16 + i][tn + j * 16 + (lane & 15)];
      }
      acc[j] = __builtin_amdgcn_wmma_f32_16x16x32_bf16(false, af, false, bf, (short)0,
                                                       acc[j], false, false);
    }
  }
#endif
  // C/D layout: vgpr v -> row tm+v (+8 for lanes 16..31), col tn + (lane&15) + 16j
  const int rBase = m0 + tm + ((lane >> 4) << 3);
  const int cBase = n0 + tn + (lane & 15);
#pragma unroll
  for (int v = 0; v < 8; ++v) {
    int gr = rBase + v;
    if (gr >= M) continue;
    size_t rowO = (size_t)gr * N;
#pragma unroll
    for (int j = 0; j < 4; ++j) {
      int c = cBase + 16 * j;
      float o = acc[j][v] + (bias ? bias[c] : 0.f);
      if (doGelu) o = gelu_f(o);
      if (outF) outF[rowO + c] = o;
      if (outB) outB[rowO + c] = (bhalf)o;
    }
  }
}

// ------------------------------------------------------------ GroupNorm stats
__global__ __launch_bounds__(256) void k_gn_stats(const float* __restrict__ y,
                                                  float* __restrict__ stats,
                                                  int perSample) {
  const int n = blockIdx.y;
  const float* base = y + (size_t)n * perSample;
  float s = 0.f, s2 = 0.f;
  for (int i = blockIdx.x * blockDim.x + threadIdx.x; i < perSample; i += gridDim.x * blockDim.x) {
    float v = base[i];
    s += v; s2 += v * v;
  }
#pragma unroll
  for (int o = 16; o >= 1; o >>= 1) {
    s  += __shfl_xor(s, o, 32);
    s2 += __shfl_xor(s2, o, 32);
  }
  if ((threadIdx.x & 31) == 0) {
    atomicAdd(&stats[n * 2 + 0], s);
    atomicAdd(&stats[n * 2 + 1], s2);
  }
}

// GN normalize + residual -> new_x (f32 + bf16)
__global__ __launch_bounds__(256) void k_merge_finalize(const float* __restrict__ y,
                                                        const float* __restrict__ stats,
                                                        const float* __restrict__ g,
                                                        const float* __restrict__ b,
                                                        const float* __restrict__ xin,
                                                        float* __restrict__ xout,
                                                        bhalf* __restrict__ xoutb) {
  const int n = blockIdx.y, l = blockIdx.x, c = threadIdx.x;
  const float inv = 1.0f / (float)(LTOK * CCH);
  float mean = stats[n * 2] * inv;
  float var = stats[n * 2 + 1] * inv - mean * mean;
  float rstd = rsqrtf(var + 1e-5f);
  size_t i = ((size_t)n * LTOK + l) * CCH + c;
  float v = (y[i] - mean) * rstd * g[c] + b[c] + xin[i];
  xout[i] = v;
  xoutb[i] = (bhalf)v;
}

// GN normalize + bilinear 52->26 downsample + residual -> new_c (f32 + bf16)
__global__ __launch_bounds__(256) void k_merge_down(const float* __restrict__ y,
                                                    const float* __restrict__ stats,
                                                    const float* __restrict__ g,
                                                    const float* __restrict__ b,
                                                    const float* __restrict__ cin,
                                                    float* __restrict__ cout,
                                                    bhalf* __restrict__ coutb) {
  const int n = blockIdx.y, s = blockIdx.x, c = threadIdx.x;
  const float inv = 1.0f / (float)(LTOK * CCH);
  float mean = stats[n * 2] * inv;
  float var = stats[n * 2 + 1] * inv - mean * mean;
  float rstd = rsqrtf(var + 1e-5f);
  const int oy = s / 26, ox = s % 26;
  const float fy = oy * (51.0f / 25.0f);
  const float fx = ox * (51.0f / 25.0f);
  int y0 = (int)fy, x0 = (int)fx;
  int y1 = min(y0 + 1, 51), x1 = min(x0 + 1, 51);
  float dy = fy - (float)y0, dx = fx - (float)x0;
  const float* yb = y + (size_t)n * LTOK * CCH + c;
  float gc = g[c], bc = b[c];
  auto val = [&](int yy, int xx) -> float {
    return (yb[(size_t)(yy * HDIM + xx) * CCH] - mean) * rstd * gc + bc;
  };
  float r0 = val(y0, x0) * (1.f - dx) + val(y0, x1) * dx;
  float r1 = val(y1, x0) * (1.f - dx) + val(y1, x1) * dx;
  size_t i = ((size_t)n * STOK + s) * CCH + c;
  float o = r0 * (1.f - dy) + r1 * dy + cin[i];
  cout[i] = o;
  coutb[i] = (bhalf)o;
}

// -------------------------------------------------- cluster: normalize centroids
__global__ __launch_bounds__(256) void k_norm_cpt(const float* __restrict__ cp,
                                                  float* __restrict__ cptn) {
  const int row = blockIdx.x * 8 + (threadIdx.x >> 5);
  const int lane = threadIdx.x & 31;
  if (row >= 64 * STOK) return;
  const int m = row / STOK, s = row % STOK;
  const int n = m >> 3, f = m & 7;
  float v = cp[((size_t)n * STOK + s) * 512 + f * 64 + lane];
  float sq = v * v;
#pragma unroll
  for (int o = 16; o >= 1; o >>= 1) sq += __shfl_xor(sq, o, 32);
  float inv = 1.0f / fmaxf(sqrtf(sq), 1e-12f);
  cptn[(size_t)row * 32 + lane] = v * inv;
}

// sim = sigmoid(a*cos + b); max/argmax over s; disp = val * cval[idx] (bf16 out)
__global__ __launch_bounds__(256) void k_sim(const float* __restrict__ xp,
                                             const float* __restrict__ cptn,
                                             const float* __restrict__ cp,
                                             const float* __restrict__ alpha_p,
                                             const float* __restrict__ beta_p,
                                             bhalf* __restrict__ disp) {
  __shared__ float sC[64][33];
  const int m = blockIdx.y;
  const int n = m >> 3, f = m & 7;
  const int l = blockIdx.x * 256 + threadIdx.x;
  const bool valid = l < LTOK;
  float xv[32];
#pragma unroll
  for (int c = 0; c < 32; ++c) xv[c] = 0.f;
  if (valid) {
    const float* xr = xp + ((size_t)n * LTOK + l) * 256 + f * 32;
    float ss = 0.f;
#pragma unroll
    for (int c = 0; c < 32; ++c) { xv[c] = xr[c]; ss += xv[c] * xv[c]; }
    float inv = 1.0f / fmaxf(sqrtf(ss), 1e-12f);
#pragma unroll
    for (int c = 0; c < 32; ++c) xv[c] *= inv;
  }
  const float alpha = alpha_p[0], beta = beta_p[0];
  float best = -1e30f;
  int bidx = 0;
  for (int s0 = 0; s0 < STOK; s0 += 64) {
    const int ns = min(64, STOK - s0);
    __syncthreads();
    for (int idx = threadIdx.x; idx < ns * 32; idx += 256) {
      int sr = idx >> 5, c = idx & 31;
      sC[sr][c] = cptn[((size_t)m * STOK + s0 + sr) * 32 + c];
    }
    __syncthreads();
    if (valid) {
      for (int sr = 0; sr < ns; ++sr) {
        float d = 0.f;
#pragma unroll
        for (int c = 0; c < 32; ++c) d += xv[c] * sC[sr][c];
        float sg = 1.0f / (1.0f + expf(-(alpha * d + beta)));
        if (sg > best) { best = sg; bidx = s0 + sr; }
      }
    }
  }
  if (valid) {
    const float* cv = cp + ((size_t)n * STOK + bidx) * 512 + f * 64 + 32;
    bhalf* dp = disp + ((size_t)n * LTOK + l) * 256 + f * 32;
#pragma unroll
    for (int c = 0; c < 32; ++c) dp[c] = (bhalf)(best * cv[c]);
  }
}

// LN(new) -> cat[...,256:512] bf16 ; copy x_bf -> cat[...,0:256]
__global__ __launch_bounds__(256) void k_ln_concat(const float* __restrict__ newv,
                                                   const float* __restrict__ g,
                                                   const float* __restrict__ b,
                                                   const bhalf* __restrict__ xbf,
                                                   bhalf* __restrict__ cat) {
  const int row = blockIdx.x * 8 + (threadIdx.x >> 5);
  const int lane = threadIdx.x & 31;
  const float* r = newv + (size_t)row * 256;
  float v[8];
  float s = 0.f, s2 = 0.f;
#pragma unroll
  for (int i = 0; i < 8; ++i) {
    v[i] = r[lane + 32 * i];
    s += v[i]; s2 += v[i] * v[i];
  }
#pragma unroll
  for (int o = 16; o >= 1; o >>= 1) {
    s += __shfl_xor(s, o, 32);
    s2 += __shfl_xor(s2, o, 32);
  }
  float mean = s * (1.0f / 256.0f);
  float var = s2 * (1.0f / 256.0f) - mean * mean;
  float rstd = rsqrtf(var + 1e-5f);
  bhalf* co = cat + (size_t)row * 512;
  const bhalf* xb = xbf + (size_t)row * 256;
#pragma unroll
  for (int i = 0; i < 8; ++i) {
    int c = lane + 32 * i;
    co[c] = xb[c];
    co[256 + c] = (bhalf)((v[i] - mean) * rstd * g[c] + b[c]);
  }
}

// LN(conv) + residual -> final output (f32)
__global__ __launch_bounds__(256) void k_ln_residual(const float* __restrict__ conv,
                                                     const float* __restrict__ g,
                                                     const float* __restrict__ b,
                                                     const float* __restrict__ xres,
                                                     float* __restrict__ out) {
  const int row = blockIdx.x * 8 + (threadIdx.x >> 5);
  const int lane = threadIdx.x & 31;
  const float* r = conv + (size_t)row * 256;
  float v[8];
  float s = 0.f, s2 = 0.f;
#pragma unroll
  for (int i = 0; i < 8; ++i) {
    v[i] = r[lane + 32 * i];
    s += v[i]; s2 += v[i] * v[i];
  }
#pragma unroll
  for (int o = 16; o >= 1; o >>= 1) {
    s += __shfl_xor(s, o, 32);
    s2 += __shfl_xor(s2, o, 32);
  }
  float mean = s * (1.0f / 256.0f);
  float var = s2 * (1.0f / 256.0f) - mean * mean;
  float rstd = rsqrtf(var + 1e-5f);
  float* o_ = out + (size_t)row * 256;
  const float* xr = xres + (size_t)row * 256;
#pragma unroll
  for (int i = 0; i < 8; ++i) {
    int c = lane + 32 * i;
    o_[c] = (v[i] - mean) * rstd * g[c] + b[c] + xr[c];
  }
}

// ------------------------------------------------- 3x3 conv as implicit WMMA GEMM
// 64 pixels x 128 oc per workgroup; flattened 9x16 tile loop, double-buffered async.
__global__ __launch_bounds__(256) void k_conv3x3(const bhalf* __restrict__ Hin,
                                                 const bhalf* __restrict__ Wp,
                                                 const float* __restrict__ bias,
                                                 float* __restrict__ outF) {
  __shared__ bhalf sA[2][64][40];
  __shared__ bhalf sB[2][32][136];
  const int tid = threadIdx.x;
  const int wave = tid >> 5, lane = tid & 31;
  const int n = blockIdx.y;
  const int p0 = blockIdx.x * 64;
  const int oc0 = blockIdx.z * 128;
  const int tm = (wave & 3) * 16, tn = (wave >> 2) * 64;
  v8f acc[4];
#pragma unroll
  for (int j = 0; j < 4; ++j) acc[j] = (v8f){0.f, 0.f, 0.f, 0.f, 0.f, 0.f, 0.f, 0.f};
  const int aRow = tm + (lane & 15);
  const int aK0  = (lane < 16) ? 0 : 8;
  const int lar = tid >> 2, lac = (tid & 3) * 8;
  const int lbr0 = tid >> 4, lbc0 = (tid & 15) * 8;
  const int lbr1 = lbr0 + 16;
  const int p = p0 + lar;
  const int py = p / HDIM, px = p % HDIM;
  // signed offsets: halo taps read slightly outside Hin image (ws-interior), zeroed below
  auto rowptr = [&](int tap) -> const bhalf* {
    int yy = py + tap / 3 - 1, xx = px + tap % 3 - 1;
    return Hin + ((long)n * LTOK + (long)(yy * HDIM + xx)) * 512 + lac;
  };
  auto rowvalid = [&](int tap) -> bool {
    int yy = py + tap / 3 - 1, xx = px + tap % 3 - 1;
    return (p < LTOK) && (yy >= 0) && (yy < HDIM) && (xx >= 0) && (xx < HDIM);
  };
  const bhalf* wbase = Wp + oc0 + lbc0;
#ifdef HAVE_ASYNC_LDS
  const unsigned loA[2]  = {lds_off_of(&sA[0][lar][lac]),  lds_off_of(&sA[1][lar][lac])};
  const unsigned loB0[2] = {lds_off_of(&sB[0][lbr0][lbc0]), lds_off_of(&sB[1][lbr0][lbc0])};
  const unsigned loB1[2] = {lds_off_of(&sB[0][lbr1][lbc0]), lds_off_of(&sB[1][lbr1][lbc0])};
  async_ld16(rowptr(0), loA[0]);
  async_ld16(wbase + (size_t)lbr0 * 256, loB0[0]);
  async_ld16(wbase + (size_t)lbr1 * 256, loB1[0]);
  int cur = 0;
  for (int t = 0; t < 144; ++t) {
    const int tap = t >> 4, kb = (t & 15) << 5;
    const int nxt = cur ^ 1;
    const bool more = (t + 1) < 144;
    if (more) {
      const int tap2 = (t + 1) >> 4, kb2 = ((t + 1) & 15) << 5;
      async_ld16(rowptr(tap2) + kb2, loA[nxt]);
      async_ld16(wbase + (size_t)(tap2 * 512 + kb2 + lbr0) * 256, loB0[nxt]);
      async_ld16(wbase + (size_t)(tap2 * 512 + kb2 + lbr1) * 256, loB1[nxt]);
    }
    if (more) async_wait3(); else async_wait0();
    if (!rowvalid(tap)) *(uint4*)&sA[cur][lar][lac] = make_uint4(0, 0, 0, 0);  // halo zero
    __syncthreads();
    const bhalf (*pA)[40]  = sA[cur];
    const bhalf (*pB)[136] = sB[cur];
    v16bh af;
#pragma unroll
    for (int i = 0; i < 8; ++i) {
      af[i]     = pA[aRow][aK0 + i];
      af[8 + i] = pA[aRow][aK0 + 16 + i];
    }
#pragma unroll
    for (int j = 0; j < 4; ++j) {
      v16bh bf;
#pragma unroll
      for (int i = 0; i < 8; ++i) {
        bf[i]     = pB[aK0 + i][tn + j * 16 + (lane & 15)];
        bf[8 + i] = pB[aK0 + 16 + i][tn + j * 16 + (lane & 15)];
      }
      acc[j] = __builtin_amdgcn_wmma_f32_16x16x32_bf16(false, af, false, bf, (short)0,
                                                       acc[j], false, false);
    }
    __syncthreads();
    cur = nxt;
  }
#else
  for (int tap = 0; tap < 9; ++tap) {
    const bool vrow = rowvalid(tap);
    const bhalf* ap = rowptr(tap);
    const bhalf* wb = Wp + (size_t)tap * 512 * 256 + oc0 + lbc0;
    for (int kb = 0; kb < 512; kb += 32) {
      __syncthreads();
      {
        uint4 va = make_uint4(0, 0, 0, 0);
        if (vrow) va = *(const uint4*)(ap + kb);
        *(uint4*)&sA[0][lar][lac] = va;
        *(uint4*)&sB[0][lbr0][lbc0] = *(const uint4*)(wb + (size_t)(kb + lbr0) * 256);
        *(uint4*)&sB[0][lbr1][lbc0] = *(const uint4*)(wb + (size_t)(kb + lbr1) * 256);
      }
      __syncthreads();
      v16bh af;
#pragma unroll
      for (int i = 0; i < 8; ++i) {
        af[i]     = sA[0][aRow][aK0 + i];
        af[8 + i] = sA[0][aRow][aK0 + 16 + i];
      }
#pragma unroll
      for (int j = 0; j < 4; ++j) {
        v16bh bf;
#pragma unroll
        for (int i = 0; i < 8; ++i) {
          bf[i]     = sB[0][aK0 + i][tn + j * 16 + (lane & 15)];
          bf[8 + i] = sB[0][aK0 + 16 + i][tn + j * 16 + (lane & 15)];
        }
        acc[j] = __builtin_amdgcn_wmma_f32_16x16x32_bf16(false, af, false, bf, (short)0,
                                                         acc[j], false, false);
      }
    }
  }
#endif
  const int rBase = tm + ((lane >> 4) << 3);
  const int cBase = oc0 + tn + (lane & 15);
#pragma unroll
  for (int v = 0; v < 8; ++v) {
    int pr = p0 + rBase + v;
    if (pr >= LTOK) continue;
    size_t rowO = ((size_t)n * LTOK + pr) * 256;
#pragma unroll
    for (int j = 0; j < 4; ++j) {
      int c = cBase + 16 * j;
      outF[rowO + c] = acc[j][v] + bias[c];
    }
  }
}

// =============================================================== host driver
static void gemm(const bhalf* A, const bhalf* B, const float* bias, float* outF,
                 bhalf* outB, int M, int N, int K, int gelu, hipStream_t s) {
  dim3 grid((M + 63) / 64, N / 128);
  k_gemm<<<grid, 256, 0, s>>>(A, B, bias, outF, outB, M, N, K, gelu);
}

extern "C" void kernel_launch(void* const* d_in, const int* in_sizes, int n_in,
                              void* d_out, int out_size, void* d_ws, size_t ws_size,
                              hipStream_t stream) {
  const float* x0_16x = (const float*)d_in[0];
  const float* x1_16x = (const float*)d_in[1];
  const float* x0_32x = (const float*)d_in[2];
  const float* x1_32x = (const float*)d_in[3];
  const float* mb_l0_w = (const float*)d_in[6];
  const float* mb_l0_b = (const float*)d_in[7];
  const float* mb_l1_w = (const float*)d_in[8];
  const float* mb_l1_b = (const float*)d_in[9];
  const float* mb_gn_g = (const float*)d_in[10];
  const float* mb_gn_b = (const float*)d_in[11];
  const float* gc_p0_w = (const float*)d_in[12];
  const float* gc_p0_b = (const float*)d_in[13];
  const float* gc_p1_w = (const float*)d_in[14];
  const float* gc_p1_b = (const float*)d_in[15];
  const float* gc_m_w = (const float*)d_in[16];
  const float* gc_m_b = (const float*)d_in[17];
  const float* gc_alpha = (const float*)d_in[18];
  const float* gc_beta = (const float*)d_in[19];
  const float* gb_n0_g = (const float*)d_in[20];
  const float* gb_n0_b = (const float*)d_in[21];
  const float* gb_lin_w = (const float*)d_in[22];
  const float* gb_lin_b = (const float*)d_in[23];
  const float* gb_conv_w = (const float*)d_in[24];
  const float* gb_conv_b = (const float*)d_in[25];
  const float* gb_n1_g = (const float*)d_in[26];
  const float* gb_n1_b = (const float*)d_in[27];

  char* ws = (char*)d_ws;
  size_t off = 0;
  auto alloc = [&](size_t bytes) -> char* {
    char* p = ws + off;
    off += (bytes + 255) & ~(size_t)255;
    return p;
  };

  // bf16 weights
  bhalf* w_l0 = (bhalf*)alloc((size_t)512 * 512 * 2);
  bhalf* w_l1 = (bhalf*)alloc((size_t)512 * 256 * 2);
  bhalf* w_p0 = (bhalf*)alloc((size_t)256 * 256 * 2);
  bhalf* w_p1 = (bhalf*)alloc((size_t)256 * 512 * 2);
  bhalf* w_m  = (bhalf*)alloc((size_t)256 * 256 * 2);
  bhalf* w_ln = (bhalf*)alloc((size_t)512 * 512 * 2);
  bhalf* w_cv = (bhalf*)alloc((size_t)9 * 512 * 256 * 2);
  // activations / scratch
  float* xA_f   = (float*)alloc((size_t)NL * CCH * 4);
  float* xB_f   = (float*)alloc((size_t)NL * CCH * 4);
  float* cA_f   = (float*)alloc((size_t)NS * CCH * 4);
  float* cB_f   = (float*)alloc((size_t)NS * CCH * 4);
  bhalf* xA_b   = (bhalf*)alloc((size_t)NL * CCH * 2);
  bhalf* xB_b   = (bhalf*)alloc((size_t)NL * CCH * 2);
  bhalf* cA_b   = (bhalf*)alloc((size_t)NS * CCH * 2);
  bhalf* cB_b   = (bhalf*)alloc((size_t)NS * CCH * 2);
  bhalf* cat_b  = (bhalf*)alloc((size_t)NL * 512 * 2);
  bhalf* h_b    = (bhalf*)alloc((size_t)NL * 512 * 2);
  float* y_f    = (float*)alloc((size_t)NL * CCH * 4);
  float* stats  = (float*)alloc(64);
  float* xp_f   = (float*)alloc((size_t)NL * 256 * 4);   // reused as "new" after sim
  float* cp_f   = (float*)alloc((size_t)NS * 512 * 4);
  float* cptn_f = (float*)alloc((size_t)64 * STOK * 32 * 4);
  bhalf* disp_b = (bhalf*)alloc((size_t)NL * 256 * 2);

  float* out0 = (float*)d_out;
  float* out1 = out0 + (size_t)NL * CCH;

  // ---- weight conversion / packing
  auto cvt = [&](const float* s, bhalf* d, int n) {
    k_f32_to_bf16<<<(n + 255) / 256, 256, 0, stream>>>(s, d, n);
  };
  cvt(mb_l0_w, w_l0, 512 * 512);
  cvt(mb_l1_w, w_l1, 512 * 256);
  cvt(gc_p0_w, w_p0, 256 * 256);
  cvt(gc_p1_w, w_p1, 256 * 512);
  cvt(gc_m_w, w_m, 256 * 256);
  cvt(gb_lin_w, w_ln, 512 * 512);
  k_pack_conv_w<<<(9 * 512 * 256 + 255) / 256, 256, 0, stream>>>(gb_conv_w, w_cv);

  // ---- merge block (x_in, c_in) -> (xo_f/xo_b, co_f/co_b)
  auto merge = [&](const float* x_in, const float* c_in, float* xo_f, bhalf* xo_b,
                   float* co_f, bhalf* co_b) {
    k_merge_cat<<<dim3(LTOK, NBATCH), 256, 0, stream>>>(x_in, c_in, cat_b);
    gemm(cat_b, w_l0, mb_l0_b, nullptr, h_b, NL, 512, 512, 1, stream);  // GELU
    gemm(h_b, w_l1, mb_l1_b, y_f, nullptr, NL, 256, 512, 0, stream);
    k_zero<<<1, 64, 0, stream>>>(stats, 16);
    k_gn_stats<<<dim3(128, NBATCH), 256, 0, stream>>>(y_f, stats, LTOK * CCH);
    k_merge_finalize<<<dim3(LTOK, NBATCH), 256, 0, stream>>>(y_f, stats, mb_gn_g, mb_gn_b,
                                                             x_in, xo_f, xo_b);
    k_merge_down<<<dim3(STOK, NBATCH), 256, 0, stream>>>(y_f, stats, mb_gn_g, mb_gn_b,
                                                         c_in, co_f, co_b);
  };
  merge(x0_16x, x0_32x, xA_f, xA_b, cA_f, cA_b);
  merge(x1_16x, x1_32x, xB_f, xB_b, cB_f, cB_b);

  // ---- global block (x_f/x_b with center c_b) -> out
  auto gblock = [&](const float* x_f, const bhalf* x_b, const bhalf* c_b, float* outp) {
    gemm(x_b, w_p0, gc_p0_b, xp_f, nullptr, NL, 256, 256, 0, stream);
    gemm(c_b, w_p1, gc_p1_b, cp_f, nullptr, NS, 512, 256, 0, stream);
    k_norm_cpt<<<(64 * STOK) / 8, 256, 0, stream>>>(cp_f, cptn_f);
    k_sim<<<dim3((LTOK + 255) / 256, 64), 256, 0, stream>>>(xp_f, cptn_f, cp_f,
                                                            gc_alpha, gc_beta, disp_b);
    float* new_f = xp_f;  // xp dead after k_sim; reuse
    gemm(disp_b, w_m, gc_m_b, new_f, nullptr, NL, 256, 256, 0, stream);
    k_ln_concat<<<NL / 8, 256, 0, stream>>>(new_f, gb_n0_g, gb_n0_b, x_b, cat_b);
    gemm(cat_b, w_ln, gb_lin_b, nullptr, h_b, NL, 512, 512, 1, stream);  // GELU
    k_conv3x3<<<dim3((LTOK + 63) / 64, NBATCH, 2), 256, 0, stream>>>(h_b, w_cv, gb_conv_b, y_f);
    k_ln_residual<<<NL / 8, 256, 0, stream>>>(y_f, gb_n1_g, gb_n1_b, x_f, outp);
  };
  gblock(xA_f, xA_b, cB_b, out0);  // x0_16x with x1_32x
  gblock(xB_f, xB_b, cA_b, out1);  // x1_16x with x0_32x
}
